// LinearRecurrentUnitFFN_14035953123925
// MI455X (gfx1250) — compile-verified, matrix operand order checked by reading, not run on hardware
//
#include <hip/hip_runtime.h>

// ---------------- problem constants ----------------
#define BQ     8
#define TT     4096
#define NN     256          // num_neurons (K of gemm1, N of gemm2)
#define HH     512          // d_hidden
#define MTOT   (BQ * TT)    // 32768 rows
#define KW     (2 * HH)     // 1024 = [re | im] packed width
#define CHUNK  64
#define NCHUNK (TT / CHUNK) // 64

typedef __attribute__((ext_vector_type(16))) __bf16 v16bf;
typedef __attribute__((ext_vector_type(8)))  __bf16 v8bf;
typedef __attribute__((ext_vector_type(8)))  float  v8f;

// ---------------- workspace layout (bytes) ----------------
// Wfrag : 64 ntiles x 8 kslabs x (32 lanes x 16) bf16  = 512 KB (WMMA-B fragment order)
// C2frag: 16 ntiles x 32 kslabs x (32 lanes x 16) bf16 = 512 KB
static constexpr size_t OFF_WF  = 0;
static constexpr size_t OFF_CF  = OFF_WF  + (size_t)64 * 8 * 512 * 2;
static constexpr size_t OFF_PAR = OFF_CF  + (size_t)16 * 32 * 512 * 2;
static constexpr size_t OFF_XBF = OFF_PAR + 32768;                       // x in bf16: MTOT x NN
static constexpr size_t OFF_BU  = OFF_XBF + (size_t)MTOT * NN * 2;       // bu: MTOT x KW f32
static constexpr size_t OFF_CE  = OFF_BU  + (size_t)MTOT * KW * 4;       // chunk ends
static constexpr size_t OFF_PF  = OFF_CE  + (size_t)BQ * NCHUNK * KW * 4;
static constexpr size_t OFF_XS  = OFF_PF  + (size_t)BQ * NCHUNK * KW * 4; // xs: MTOT x KW bf16
static constexpr size_t WS_NEED = OFF_XS  + (size_t)MTOT * KW * 2;

// params (f32): [0:H) lam_re  [H:2H) lam_im  [2H:3H) gamma  [3H:4H) lam^CHUNK re  [4H:5H) im

// K index inside a 32-deep slab for fragment element i of a lane (16-bit B layout,
// mirrored from the documented A layout): halves {0..7,16..23} / {8..15,24..31}
__device__ __forceinline__ int frag_k(int lane, int i) {
  return ((lane >> 4) ? 8 : 0) + ((i < 8) ? i : (16 + (i - 8)));
}

// ================= prep: fragment-packed bf16 weights + per-h params =================
__global__ __launch_bounds__(256) void lru_prep(
    const float* __restrict__ Bre, const float* __restrict__ Bim,
    const float* __restrict__ Cre, const float* __restrict__ Cim,
    const float* __restrict__ nu,  const float* __restrict__ theta,
    const float* __restrict__ gamma_log,
    __bf16* __restrict__ Wfrag, __bf16* __restrict__ C2frag,
    float* __restrict__ params)
{
  const int idx = blockIdx.x * 256 + threadIdx.x;   // 0 .. 262143
  {
    // Wfrag: ntile(64) | kslab(8) | lane(32) | i(16);  W is K=NN rows x KW cols
    const int i = idx & 15, lane = (idx >> 4) & 31, ks = (idx >> 9) & 7, ntile = idx >> 12;
    const int k = ks * 32 + frag_k(lane, i);                 // 0..255
    const int j = ntile * 16 + (lane & 15);                  // 0..1023
    const float v = (j < HH) ? Bre[(size_t)k * HH + j] : Bim[(size_t)k * HH + (j - HH)];
    Wfrag[idx] = (__bf16)v;
  }
  {
    // C2frag: ntile(16) | kslab(32) | lane(32) | i(16);  C2 is K=KW rows x NN cols
    const int i = idx & 15, lane = (idx >> 4) & 31, ks = (idx >> 9) & 31, ntile = idx >> 14;
    const int k = ks * 32 + frag_k(lane, i);                 // 0..1023 (= h2)
    const int n = ntile * 16 + (lane & 15);                  // 0..255
    const float v = (k < HH) ? Cre[(size_t)k * NN + n] : -Cim[(size_t)(k - HH) * NN + n];
    C2frag[idx] = (__bf16)v;
  }
  if (idx < HH) {
    const float mag = __expf(-__expf(nu[idx]));   // |lambda|
    const float th  = theta[idx];
    const float lre = mag * __cosf(th);
    const float lim = mag * __sinf(th);
    params[idx]          = lre;
    params[HH + idx]     = lim;
    params[2 * HH + idx] = __expf(gamma_log[idx]);
    float ar = lre, ai = lim;                      // lam^64 = 6 complex squarings
    #pragma unroll
    for (int s = 0; s < 6; ++s) { float nr = ar * ar - ai * ai; float ni = 2.f * ar * ai; ar = nr; ai = ni; }
    params[3 * HH + idx] = ar;
    params[4 * HH + idx] = ai;
  }
}

// ================= convert x to bf16 once =================
__global__ __launch_bounds__(256) void lru_cvt_x(
    const float* __restrict__ x, __bf16* __restrict__ xbf)
{
  const size_t idx = ((size_t)blockIdx.x * 256 + threadIdx.x) * 8;
  const float4 a = *(const float4*)(x + idx);
  const float4 b = *(const float4*)(x + idx + 4);
  v8bf o;
  o[0]=(__bf16)a.x; o[1]=(__bf16)a.y; o[2]=(__bf16)a.z; o[3]=(__bf16)a.w;
  o[4]=(__bf16)b.x; o[5]=(__bf16)b.y; o[6]=(__bf16)b.z; o[7]=(__bf16)b.w;
  *(v8bf*)(xbf + idx) = o;
}

// ---- assemble a 16-element bf16 WMMA fragment from two 16B chunks ----
__device__ __forceinline__ v16bf load_frag_pair(const __bf16* lo_p, const __bf16* hi_p) {
  const v8bf lo = *(const v8bf*)lo_p;
  const v8bf hi = *(const v8bf*)hi_p;
  v16bf a;
  #pragma unroll
  for (int i = 0; i < 8; ++i) { a[i] = lo[i]; a[8 + i] = hi[i]; }
  return a;
}
__device__ __forceinline__ v16bf lds_frag(const __bf16* p) {   // 32B contiguous in LDS
  const v8bf lo = *(const v8bf*)p;
  const v8bf hi = *(const v8bf*)(p + 8);
  v16bf a;
  #pragma unroll
  for (int i = 0; i < 8; ++i) { a[i] = lo[i]; a[8 + i] = hi[i]; }
  return a;
}

// ================= GEMM1: bu = gamma * (x @ W) =================
// Block = 8 waves sharing one 64-wide N group; B panel (32 KB) staged in LDS once.
__global__ __launch_bounds__(256) void lru_gemm1(
    const __bf16* __restrict__ xbf, const __bf16* __restrict__ Wfrag,
    const float* __restrict__ params, float* __restrict__ bu)
{
  __shared__ __bf16 Bsh[4 * 8 * 512];                       // 32 KB
  const int tid  = threadIdx.x;
  const int lane = tid & 31;
  const int wv   = tid >> 5;
  const int mb   = blockIdx.x >> 4;                         // 256 m-blocks
  const int ng   = blockIdx.x & 15;                         // 16 n-groups of 64
  const int tm   = (mb * 8 + wv) * 16;
  const int mrow = tm + (lane & 15);
  const int koff = (lane >> 4) ? 8 : 0;

  // cooperative LDS fill: 16384 bf16 contiguous from Wfrag
  {
    const __bf16* src = Wfrag + (size_t)(ng * 4) * (8 * 512);
    #pragma unroll
    for (int r = 0; r < 8; ++r) {
      const int e = (r * 256 + tid) * 8;
      *(v8bf*)(Bsh + e) = *(const v8bf*)(src + e);
    }
  }
  __syncthreads();

  v8f acc[4] = {v8f{}, v8f{}, v8f{}, v8f{}};
  for (int ks = 0; ks < NN / 32; ++ks) {
    const __bf16* ap = xbf + (size_t)mrow * NN + ks * 32 + koff;
    const v16bf a = load_frag_pair(ap, ap + 16);
    if (ks + 1 < NN / 32) __builtin_prefetch(ap + 32, 0, 0);   // global_prefetch next A
    #pragma unroll
    for (int t = 0; t < 4; ++t) {
      const v16bf b = lds_frag(Bsh + (t * 8 + ks) * 512 + lane * 16);
      acc[t] = __builtin_amdgcn_wmma_f32_16x16x32_bf16(false, a, false, b, (short)0, acc[t], false, false);
    }
  }
  const int mbase = tm + 8 * (lane >> 4);
  #pragma unroll
  for (int t = 0; t < 4; ++t) {
    const int ncol = (ng * 4 + t) * 16 + (lane & 15);
    const float g = params[2 * HH + (ncol & (HH - 1))];
    float* op = bu + (size_t)mbase * KW + ncol;
    #pragma unroll
    for (int r = 0; r < 8; ++r) op[(size_t)r * KW] = acc[t][r] * g;
  }
}

// ================= scan phase 1: chunk-local end states =================
__global__ __launch_bounds__(256) void lru_scan1(
    const float* __restrict__ bu, const float* __restrict__ params,
    float* __restrict__ chunkend)
{
  const int tid = blockIdx.x * 256 + threadIdx.x;  // 262144 = B*NCHUNK*H
  const int h = tid & (HH - 1);
  const int rest = tid >> 9;
  const int c = rest & (NCHUNK - 1);
  const int b = rest >> 6;
  const float lre = params[h], lim = params[HH + h];
  float cr = 0.f, ci = 0.f;
  const float* p = bu + (size_t)(b * TT + c * CHUNK) * KW + h;
  for (int i = 0; i < CHUNK; ++i) {
    const float br = p[0], bi = p[HH];
    const float nr = lre * cr - lim * ci + br;
    const float ni = lre * ci + lim * cr + bi;
    cr = nr; ci = ni;
    p += KW;
  }
  const size_t o = (size_t)(b * NCHUNK + c) * KW + h;
  chunkend[o] = cr; chunkend[o + HH] = ci;
}

// ================= scan phase 2: prefix over chunk summaries =================
__global__ __launch_bounds__(256) void lru_scan2(
    const float* __restrict__ chunkend, const float* __restrict__ params,
    float* __restrict__ prefix)
{
  const int tid = blockIdx.x * 256 + threadIdx.x;  // 4096 = B*H
  const int h = tid & (HH - 1);
  const int b = tid >> 9;
  const float Lr = params[3 * HH + h], Li = params[4 * HH + h];  // lam^CHUNK
  float pr = 0.f, pi = 0.f;
  for (int c = 0; c < NCHUNK; ++c) {
    const size_t o = (size_t)(b * NCHUNK + c) * KW + h;
    prefix[o] = pr; prefix[o + HH] = pi;
    const float er = chunkend[o], ei = chunkend[o + HH];
    const float nr = Lr * pr - Li * pi + er;
    const float ni = Lr * pi + Li * pr + ei;
    pr = nr; pi = ni;
  }
}

// ================= scan phase 3: replay with carry, emit xs as bf16 =================
__global__ __launch_bounds__(256) void lru_scan3(
    const float* __restrict__ bu, const float* __restrict__ params,
    const float* __restrict__ prefix, __bf16* __restrict__ xsbf)
{
  const int tid = blockIdx.x * 256 + threadIdx.x;  // 262144
  const int h = tid & (HH - 1);
  const int rest = tid >> 9;
  const int c = rest & (NCHUNK - 1);
  const int b = rest >> 6;
  const float lre = params[h], lim = params[HH + h];
  const size_t po = (size_t)(b * NCHUNK + c) * KW + h;
  float cr = prefix[po], ci = prefix[po + HH];
  const float* p = bu + (size_t)(b * TT + c * CHUNK) * KW + h;
  __bf16* q = xsbf + (size_t)(b * TT + c * CHUNK) * KW + h;
  for (int i = 0; i < CHUNK; ++i) {
    const float br = p[0], bi = p[HH];
    const float nr = lre * cr - lim * ci + br;
    const float ni = lre * ci + lim * cr + bi;
    cr = nr; ci = ni;
    q[0]  = (__bf16)cr;
    q[HH] = (__bf16)ci;
    p += KW; q += KW;
  }
}

// ================= GEMM2: y = xs @ [Cre; -Cim] + D*u =================
// Block = 8 waves sharing one 64-wide N group; B panel staged in LDS in 2 x 64 KB stages.
__global__ __launch_bounds__(256) void lru_gemm2(
    const __bf16* __restrict__ xsbf, const __bf16* __restrict__ C2frag,
    const float* __restrict__ x, const float* __restrict__ D,
    float* __restrict__ y)
{
  __shared__ __bf16 Bsh[4 * 16 * 512];                      // 64 KB (half-K stage)
  const int tid  = threadIdx.x;
  const int lane = tid & 31;
  const int wv   = tid >> 5;
  const int mb   = blockIdx.x >> 2;                         // 256 m-blocks
  const int ng   = blockIdx.x & 3;                          // 4 n-groups of 64
  const int tm   = (mb * 8 + wv) * 16;
  const int mrow = tm + (lane & 15);
  const int koff = (lane >> 4) ? 8 : 0;

  v8f acc[4] = {v8f{}, v8f{}, v8f{}, v8f{}};
  for (int s = 0; s < 2; ++s) {
    __syncthreads();                 // previous stage's reads complete
    // cooperative LDS fill: 4 ntiles x 16 kslabs x 512 bf16 (per-ntile contiguous runs)
    #pragma unroll
    for (int r = 0; r < 16; ++r) {
      const int e   = (r * 256 + tid) * 8;      // 0 .. 32767 step 8
      const int t   = e >> 13;                  // ntile within group
      const int rem = e & 8191;                 // 16 kslabs x 512
      const __bf16* src = C2frag + ((size_t)((ng * 4 + t) * 32 + s * 16) * 512) + rem;
      *(v8bf*)(Bsh + e) = *(const v8bf*)(src);
    }
    __syncthreads();
    for (int kk = 0; kk < 16; ++kk) {
      const int ks = s * 16 + kk;
      const __bf16* ap = xsbf + (size_t)mrow * KW + ks * 32 + koff;
      const v16bf a = load_frag_pair(ap, ap + 16);
      if (ks + 1 < KW / 32) __builtin_prefetch(ap + 32, 0, 0);
      #pragma unroll
      for (int t = 0; t < 4; ++t) {
        const v16bf b = lds_frag(Bsh + (t * 16 + kk) * 512 + lane * 16);
        acc[t] = __builtin_amdgcn_wmma_f32_16x16x32_bf16(false, a, false, b, (short)0, acc[t], false, false);
      }
    }
  }
  const int mbase = tm + 8 * (lane >> 4);
  #pragma unroll
  for (int t = 0; t < 4; ++t) {
    const int ncol = (ng * 4 + t) * 16 + (lane & 15);
    const float d = D[ncol];
    #pragma unroll
    for (int r = 0; r < 8; ++r) {
      const size_t o = (size_t)(mbase + r) * NN + ncol;
      y[o] = acc[t][r] + d * x[o];
    }
  }
}

// ================= launcher =================
extern "C" void kernel_launch(void* const* d_in, const int* in_sizes, int n_in,
                              void* d_out, int out_size, void* d_ws, size_t ws_size,
                              hipStream_t stream) {
  const float* x   = (const float*)d_in[0];
  const float* Bre = (const float*)d_in[1];
  const float* Bim = (const float*)d_in[2];
  const float* Cre = (const float*)d_in[3];
  const float* Cim = (const float*)d_in[4];
  const float* D   = (const float*)d_in[5];
  const float* nu  = (const float*)d_in[6];
  const float* th  = (const float*)d_in[7];
  const float* gl  = (const float*)d_in[8];

  if (ws_size < WS_NEED) return;  // workspace contract violated; nothing safe to do

  char* ws = (char*)d_ws;
  __bf16* Wfrag   = (__bf16*)(ws + OFF_WF);
  __bf16* C2frag  = (__bf16*)(ws + OFF_CF);
  float*  params  = (float*)(ws + OFF_PAR);
  __bf16* xbf     = (__bf16*)(ws + OFF_XBF);
  float*  bu      = (float*)(ws + OFF_BU);
  float*  chunkend= (float*)(ws + OFF_CE);
  float*  prefix  = (float*)(ws + OFF_PF);
  __bf16* xsbf    = (__bf16*)(ws + OFF_XS);
  float*  y       = (float*)d_out;

  lru_prep <<<1024, 256, 0, stream>>>(Bre, Bim, Cre, Cim, nu, th, gl, Wfrag, C2frag, params);
  lru_cvt_x<<<4096, 256, 0, stream>>>(x, xbf);                 // 8.4M elems, 8/thread
  lru_gemm1<<<4096, 256, 0, stream>>>(xbf, Wfrag, params, bu); // 256 mb x 16 ng
  lru_scan1<<<1024, 256, 0, stream>>>(bu, params, chunkend);
  lru_scan2<<<16,   256, 0, stream>>>(chunkend, params, prefix);
  lru_scan3<<<1024, 256, 0, stream>>>(bu, params, prefix, xsbf);
  lru_gemm2<<<1024, 256, 0, stream>>>(xsbf, C2frag, x, D, y);  // 256 mb x 4 ng
}
